// MultiHeadAttention_249108103229
// MI455X (gfx1250) — matrix operand order repaired and checked by reading, not compiled, not run on hardware
//
#include <hip/hip_runtime.h>
#include <hip/hip_bf16.h>

// ---------------------------------------------------------------------------
// MI455X (gfx1250) headless attention, bf16 WMMA + async global->LDS DMA.
//   pre-pass: f32 -> bf16 for activations (query/key/value) and weights.
//   q=Aq@Wq^T+bq ; k=Ak@Wk^T+bk (bf16) ; Vt = (Av@Wv^T+bv)^T (bf16, [D, B*S])
//   Sc = (q@k^T)/sqrt(128) (f32) -> P = softmax(Sc) (bf16, in place)
//   X = P@Vt^T (bf16) ; out = X@Wo^T + bo (f32)
// One templated 128x128x32 GEMM: double-buffered LDS, async-to-LDS staging,
// 8 wave32/block, 8 v_wmma_f32_16x16x32_bf16 per wave per K-step.
// ---------------------------------------------------------------------------

#define USE_ASYNC_LDS 1

typedef __attribute__((ext_vector_type(16))) __bf16 v16bf;
typedef __attribute__((ext_vector_type(8)))  __bf16 v8bf;
typedef __attribute__((ext_vector_type(8)))  float  v8f;

union Frag { v16bf v; v8bf h[2]; };

constexpr int BM = 128, BN = 128, BK = 32, NTHREADS = 256;

#if USE_ASYNC_LDS
// One lane moves 16B global -> LDS via the async DMA path (ASYNCcnt-tracked).
// Low 32 bits of a generic pointer to __shared__ are the LDS byte offset
// (flat-LDS aperture lives entirely in the high 32 bits on gfx1250).
__device__ __forceinline__ void async_ld16(__bf16* ldst, const __bf16* gsrc) {
  unsigned loff = (unsigned)(unsigned long long)ldst;
  unsigned long long ga = (unsigned long long)gsrc;
  asm volatile("global_load_async_to_lds_b128 %0, %1, off"
               :: "v"(loff), "v"(ga) : "memory");
}
__device__ __forceinline__ void wait_async0() {
  asm volatile("s_wait_asynccnt 0x0" ::: "memory");
}
#endif

// Stage one 128x32 bf16 tile (row-major, ld = row stride) into LDS [r][k].
// Each of 256 threads moves 32 bytes.
__device__ __forceinline__ void stage_tile(__bf16* lbase, const __bf16* gbase,
                                           int ld, int tid) {
  const int r = tid >> 1, seg = tid & 1;
  const __bf16* src = gbase + (long long)r * ld + seg * 16;
  __bf16* dst = lbase + r * BK + seg * 16;
#if USE_ASYNC_LDS
  async_ld16(dst, src);
  async_ld16(dst + 8, src + 8);
#else
  ((v8bf*)dst)[0] = ((const v8bf*)src)[0];
  ((v8bf*)dst)[1] = ((const v8bf*)src)[1];
#endif
}

// C = scale*(A @ B^T) + bias.  A:[M,K] bf16, B:[N,K] bf16 (both row-major).
// STORE_TRANS: write C transposed ([N,M], ldc = M-stride, packed b128 stores).
// grid: x = M/BM, y = N/BN, z = batch (per-operand strides).
template <typename TOut, bool STORE_TRANS, bool HAS_BIAS>
__global__ __launch_bounds__(NTHREADS)
void wmma_gemm(const __bf16* __restrict__ A, int lda, long long strideA,
               const __bf16* __restrict__ Bsrc, int ldb, long long strideB,
               TOut* __restrict__ C, int ldc, long long strideC,
               const float* __restrict__ bias, int K, float scale) {
  __shared__ alignas(16) __bf16 As[2 * BM * BK];   // double-buffered [m][k]
  __shared__ alignas(16) __bf16 Bs[2 * BN * BK];   // double-buffered [n][k]

  const int tid  = threadIdx.x;
  const int lane = tid & 31;
  const int w    = tid >> 5;     // 0..7
  const int waveRow = w >> 1;    // 0..3 (32 rows each)
  const int waveCol = w & 1;     // 0..1 (64 cols each)
  const int lr = lane & 15;
  const int hk = lane >> 4;

  const long long z = blockIdx.z;
  A    += z * strideA;
  Bsrc += z * strideB;
  C    += z * strideC;

  const int rowBase = blockIdx.x * BM;
  const int colBase = blockIdx.y * BN;
  const __bf16* Ag = A + (long long)rowBase * lda;
  const __bf16* Bg = Bsrc + (long long)colBase * ldb;

  v8f zero = {};
  v8f acc[2][4];
  #pragma unroll
  for (int i = 0; i < 2; ++i)
    #pragma unroll
    for (int j = 0; j < 4; ++j) acc[i][j] = zero;

  // prime buffer 0
  stage_tile(As, Ag, lda, tid);
  stage_tile(Bs, Bg, ldb, tid);

  int cur = 0;
  for (int k0 = 0; k0 < K; k0 += BK, cur ^= 1) {
#if USE_ASYNC_LDS
    wait_async0();              // this wave's DMA into buf[cur] done
#endif
    __syncthreads();            // everyone's staging of buf[cur] visible
    if (k0 + BK < K) {          // kick off DMA for next slab into buf[cur^1]
      stage_tile(As + (cur ^ 1) * BM * BK, Ag + k0 + BK, lda, tid);
      stage_tile(Bs + (cur ^ 1) * BN * BK, Bg + k0 + BK, ldb, tid);
    }
    const __bf16* Ab = As + cur * BM * BK;
    const __bf16* Bb = Bs + cur * BN * BK;

    Frag a[2], b[4];
    #pragma unroll
    for (int mi = 0; mi < 2; ++mi) {       // A 16x32: lane<16 K0-7|K16-23
      const __bf16* p = &Ab[(waveRow * 32 + mi * 16 + lr) * BK + hk * 8];
      a[mi].h[0] = *(const v8bf*)p;
      a[mi].h[1] = *(const v8bf*)(p + 16);
    }
    #pragma unroll
    for (int ni = 0; ni < 4; ++ni) {       // B 32x16: lane<16 K0-7|K16-23
      const __bf16* q = &Bb[(waveCol * 64 + ni * 16 + lr) * BK + hk * 8];
      b[ni].h[0] = *(const v8bf*)q;
      b[ni].h[1] = *(const v8bf*)(q + 16);
    }
    #pragma unroll
    for (int mi = 0; mi < 2; ++mi)
      #pragma unroll
      for (int ni = 0; ni < 4; ++ni)
        acc[mi][ni] = __builtin_amdgcn_wmma_f32_16x16x32_bf16(
            false, a[mi].v, false, b[ni].v, (short)0, acc[mi][ni], false, false);
  }

  // epilogue: C layout -> lane holds N=lr, M = hk*8 + j (j = acc VGPR index)
  #pragma unroll
  for (int ni = 0; ni < 4; ++ni) {
    const int col = colBase + waveCol * 64 + ni * 16 + lr;
    float bv = 0.0f;
    if constexpr (HAS_BIAS) bv = bias[col];
    #pragma unroll
    for (int mi = 0; mi < 2; ++mi) {
      const int rowTop = rowBase + waveRow * 32 + mi * 16 + hk * 8;
      if constexpr (STORE_TRANS) {
        // 8 consecutive M for one N: contiguous in C^T -> one b128 store
        v8bf pk = { (__bf16)(acc[mi][ni][0] * scale + bv),
                    (__bf16)(acc[mi][ni][1] * scale + bv),
                    (__bf16)(acc[mi][ni][2] * scale + bv),
                    (__bf16)(acc[mi][ni][3] * scale + bv),
                    (__bf16)(acc[mi][ni][4] * scale + bv),
                    (__bf16)(acc[mi][ni][5] * scale + bv),
                    (__bf16)(acc[mi][ni][6] * scale + bv),
                    (__bf16)(acc[mi][ni][7] * scale + bv) };
        *(v8bf*)&C[(long long)col * ldc + rowTop] = pk;
      } else {
        #pragma unroll
        for (int j = 0; j < 8; ++j) {
          float v = acc[mi][ni][j] * scale + bv;
          C[(long long)(rowTop + j) * ldc + col] = (TOut)v;
        }
      }
    }
  }
}

// elementwise f32 -> bf16 (n multiple of 2048)
__global__ __launch_bounds__(256)
void cvt_bf16_kernel(const float* __restrict__ in, __bf16* __restrict__ out) {
  const long long i = ((long long)blockIdx.x * 256 + threadIdx.x) * 8;
  float4 a = ((const float4*)(in + i))[0];
  float4 b = ((const float4*)(in + i))[1];
  v8bf o = { (__bf16)a.x, (__bf16)a.y, (__bf16)a.z, (__bf16)a.w,
             (__bf16)b.x, (__bf16)b.y, (__bf16)b.z, (__bf16)b.w };
  *(v8bf*)(out + i) = o;
}

// Row softmax over ncols f32; writes normalized bf16 P into the head of the
// same row (safe: all reads precede the block barriers).
__global__ __launch_bounds__(256)
void softmax_row_kernel(float* __restrict__ Sc, int ncols) {
  const long long row = blockIdx.x;
  float* rp = Sc + row * (long long)ncols;
  const int tid = threadIdx.x;  // 256 threads, 8 elements each
  float4 x0 = ((const float4*)rp)[tid * 2];
  float4 x1 = ((const float4*)rp)[tid * 2 + 1];

  float m = fmaxf(fmaxf(fmaxf(x0.x, x0.y), fmaxf(x0.z, x0.w)),
                  fmaxf(fmaxf(x1.x, x1.y), fmaxf(x1.z, x1.w)));
  __shared__ float red[256];
  red[tid] = m; __syncthreads();
  for (int s = 128; s > 0; s >>= 1) {
    if (tid < s) red[tid] = fmaxf(red[tid], red[tid + s]);
    __syncthreads();
  }
  m = red[0];
  __syncthreads();

  float e[8];
  e[0] = __expf(x0.x - m); e[1] = __expf(x0.y - m);
  e[2] = __expf(x0.z - m); e[3] = __expf(x0.w - m);
  e[4] = __expf(x1.x - m); e[5] = __expf(x1.y - m);
  e[6] = __expf(x1.z - m); e[7] = __expf(x1.w - m);
  float s8 = ((e[0] + e[1]) + (e[2] + e[3])) + ((e[4] + e[5]) + (e[6] + e[7]));
  red[tid] = s8; __syncthreads();
  for (int s = 128; s > 0; s >>= 1) {
    if (tid < s) red[tid] += red[tid + s];
    __syncthreads();
  }
  const float inv = 1.0f / red[0];

  v8bf o = { (__bf16)(e[0] * inv), (__bf16)(e[1] * inv),
             (__bf16)(e[2] * inv), (__bf16)(e[3] * inv),
             (__bf16)(e[4] * inv), (__bf16)(e[5] * inv),
             (__bf16)(e[6] * inv), (__bf16)(e[7] * inv) };
  ((v8bf*)rp)[tid] = o;
}

extern "C" void kernel_launch(void* const* d_in, const int* in_sizes, int n_in,
                              void* d_out, int out_size, void* d_ws, size_t ws_size,
                              hipStream_t stream) {
  (void)in_sizes; (void)n_in; (void)out_size; (void)ws_size;
  const float* query = (const float*)d_in[0];
  const float* key_  = (const float*)d_in[1];
  const float* value = (const float*)d_in[2];
  const float* Wq = (const float*)d_in[3]; const float* bq = (const float*)d_in[4];
  const float* Wk = (const float*)d_in[5]; const float* bk = (const float*)d_in[6];
  const float* Wv = (const float*)d_in[7]; const float* bv = (const float*)d_in[8];
  const float* Wo = (const float*)d_in[9]; const float* bo = (const float*)d_in[10];
  float* out = (float*)d_out;

  const int B = 8, S = 2048, D = 1024;
  const long long MS = (long long)B * S;   // 16384 flat rows
  const long long ND = MS * D;             // 16M activation elements
  const long long NW = (long long)D * D;   // 1M weight elements

  // workspace carve (bf16 unless noted)
  __bf16* Aq  = (__bf16*)d_ws;        // bf16 query
  __bf16* Ak  = Aq + ND;              // bf16 key
  __bf16* Av  = Ak + ND;              // bf16 value
  __bf16* Wqb = Av + ND;              // bf16 weights
  __bf16* Wkb = Wqb + NW;
  __bf16* Wvb = Wkb + NW;
  __bf16* Wob = Wvb + NW;
  __bf16* Qb  = Wob + NW;             // projected q
  __bf16* Kb  = Qb + ND;              // projected k
  __bf16* Vt  = Kb + ND;              // projected v, transposed [D, MS]
  float*  Sc  = (float*)(Vt + ND);    // scores f32 (reused as bf16 P)
  __bf16* Xb  = (__bf16*)(Sc + (long long)B * S * S);

  const dim3 blk(NTHREADS);

  // 0) f32 -> bf16 conversions
  cvt_bf16_kernel<<<dim3((unsigned)(ND / 2048)), blk, 0, stream>>>(query, Aq);
  cvt_bf16_kernel<<<dim3((unsigned)(ND / 2048)), blk, 0, stream>>>(key_, Ak);
  cvt_bf16_kernel<<<dim3((unsigned)(ND / 2048)), blk, 0, stream>>>(value, Av);
  cvt_bf16_kernel<<<dim3((unsigned)(NW / 2048)), blk, 0, stream>>>(Wq, Wqb);
  cvt_bf16_kernel<<<dim3((unsigned)(NW / 2048)), blk, 0, stream>>>(Wk, Wkb);
  cvt_bf16_kernel<<<dim3((unsigned)(NW / 2048)), blk, 0, stream>>>(Wv, Wvb);
  cvt_bf16_kernel<<<dim3((unsigned)(NW / 2048)), blk, 0, stream>>>(Wo, Wob);

  // 1) projections: [16384,1024] x [1024,1024]^T + bias
  const dim3 gproj((unsigned)(MS / BM), D / BN, 1);
  wmma_gemm<__bf16, false, true><<<gproj, blk, 0, stream>>>(
      Aq, D, 0, Wqb, D, 0, Qb, D, 0, bq, D, 1.0f);
  wmma_gemm<__bf16, false, true><<<gproj, blk, 0, stream>>>(
      Ak, D, 0, Wkb, D, 0, Kb, D, 0, bk, D, 1.0f);
  // V projection stored transposed: Vt[n][flat_row], ldc = MS
  wmma_gemm<__bf16, true, true><<<gproj, blk, 0, stream>>>(
      Av, D, 0, Wvb, D, 0, Vt, (int)MS, 0, bv, D, 1.0f);

  // 2) scores = (Q @ K^T)/sqrt(128), per batch -> f32
  const dim3 gsc(S / BM, S / BN, B);
  const float scl = 0.08838834764831845f;  // 1/sqrt(128)
  wmma_gemm<float, false, false><<<gsc, blk, 0, stream>>>(
      Qb, D, (long long)S * D, Kb, D, (long long)S * D,
      Sc, S, (long long)S * S, nullptr, D, scl);

  // 3) row softmax -> bf16 P in place (row stride 2*S bf16 elements)
  softmax_row_kernel<<<dim3((unsigned)(B * S)), blk, 0, stream>>>(Sc, S);

  // 4) X = P @ V: B-operand is Vt [N=D rows, K=S cols], per-batch col offset
  const dim3 gpv(S / BM, D / BN, B);
  wmma_gemm<__bf16, false, false><<<gpv, blk, 0, stream>>>(
      (const __bf16*)Sc, 2 * S, (long long)S * 2 * S,
      Vt, (int)MS, (long long)S,
      Xb, D, (long long)S * D, nullptr, S, 1.0f);

  // 5) out = X @ Wo^T + bo -> f32
  wmma_gemm<float, false, true><<<gproj, blk, 0, stream>>>(
      Xb, D, 0, Wob, D, 0, out, D, 0, bo, D, 1.0f);
}
